// SelfAttention_35519379538488
// MI455X (gfx1250) — matrix operand order repaired and checked by reading, not compiled, not run on hardware
//
#include <hip/hip_runtime.h>
#include <hip/hip_bf16.h>

// Problem constants (from reference): B=4, C=64, H=W=64 -> N=4096, d=C/8=8.
#define BATCH 4
#define CCH   64
#define CD    8
#define NPIX  4096
#define DK    32   // K-dim padded 8 -> 32 for v_wmma_f32_16x16x32_bf16

typedef __attribute__((ext_vector_type(16))) __bf16 v16bf;
typedef __attribute__((ext_vector_type(8)))  __bf16 v8bf;
typedef __attribute__((ext_vector_type(8)))  float  v8f;

union BF16Frag {
  v16bf v;
  v8bf  h[2];
  unsigned short us[16];
};

__device__ inline unsigned short f32_to_bf16(float f) {
  unsigned int u = __float_as_uint(f);
  u += 0x7fffu + ((u >> 16) & 1u);   // round-to-nearest-even
  return (unsigned short)(u >> 16);
}

// ---------------------------------------------------------------------------
// Kernel 1: per-pixel 1x1-conv projections q,k (bf16, K-padded to 32) and v.
// Qbf/Kbf layout: [b][n][32] bf16 (elements 8..31 zero). Vbf: [b][c][n] bf16.
// ---------------------------------------------------------------------------
__global__ __launch_bounds__(256) void proj_kernel(
    const float* __restrict__ x,
    const float* __restrict__ Wq, const float* __restrict__ bq,
    const float* __restrict__ Wk, const float* __restrict__ bk,
    const float* __restrict__ Wv, const float* __restrict__ bv,
    unsigned short* __restrict__ Qbf, unsigned short* __restrict__ Kbf,
    unsigned short* __restrict__ Vbf)
{
  const int t = blockIdx.x * blockDim.x + threadIdx.x;   // one thread per pixel
  const int b = t / NPIX;
  const int n = t % NPIX;

  const float* xp = x + (size_t)b * CCH * NPIX + n;
  float xv[CCH];
  #pragma unroll
  for (int c = 0; c < CCH; ++c) xv[c] = xp[(size_t)c * NPIX];

  unsigned short* qrow = Qbf + (size_t)(b * NPIX + n) * DK;
  unsigned short* krow = Kbf + (size_t)(b * NPIX + n) * DK;
  #pragma unroll
  for (int d = 0; d < CD; ++d) {
    float qa = bq[d], ka = bk[d];
    #pragma unroll
    for (int c = 0; c < CCH; ++c) {
      qa = fmaf(Wq[d * CCH + c], xv[c], qa);
      ka = fmaf(Wk[d * CCH + c], xv[c], ka);
    }
    qrow[d] = f32_to_bf16(qa);
    krow[d] = f32_to_bf16(ka);
  }
  #pragma unroll
  for (int d = CD; d < DK; ++d) { qrow[d] = 0; krow[d] = 0; }

  for (int co = 0; co < CCH; ++co) {
    float va = bv[co];
    #pragma unroll
    for (int c = 0; c < CCH; ++c) va = fmaf(Wv[co * CCH + c], xv[c], va);
    Vbf[(size_t)(b * CCH + co) * NPIX + n] = f32_to_bf16(va);
  }
}

// ---------------------------------------------------------------------------
// Kernel 2: per-key softmax stats (softmax is over the QUERY axis).
// Each wave owns 16 key columns; streams all queries in 16-row WMMA tiles and
// maintains the online (max, sum-exp) per column.
// C/D layout: lane holds column N=lane&15, rows M = r (+8 for lanes 16..31).
// ---------------------------------------------------------------------------
__global__ __launch_bounds__(128) void stats_kernel(
    const unsigned short* __restrict__ Qbf,
    const unsigned short* __restrict__ Kbf,
    float* __restrict__ maxbuf, float* __restrict__ sumbuf)
{
  const int lane = threadIdx.x & 31;
  const int wave = threadIdx.x >> 5;
  const int b    = blockIdx.x / (NPIX / 64);
  const int mgrp = blockIdx.x % (NPIX / 64);
  const int m_base = mgrp * 64 + wave * 16;

  // Loop-invariant B-fragment: 16 key columns. B 32x16 bf16 layout: lanes 0-15
  // carry K in {0..7,16..23} (our real 8 dims + zero pad); lanes 16-31 carry K in {8..15,24..31} = 0.
  BF16Frag kb;
  #pragma unroll
  for (int i = 0; i < 16; ++i) kb.us[i] = 0;
  if (lane < 16)
    kb.v = *(const v16bf*)(Kbf + (size_t)(b * NPIX + m_base + lane) * DK);

  float run_max = -1e30f, run_sum = 0.f;

  for (int nt = 0; nt < NPIX / 16; ++nt) {
    // A 16x32 bf16 layout: lanes 0-15 row M, K in {0..7,16..23}; lanes 16-31
    // K in {8..15,24..31} which is all zero padding for us.
    BF16Frag qa;
    #pragma unroll
    for (int i = 0; i < 16; ++i) qa.us[i] = 0;
    if (lane < 16)
      qa.v = *(const v16bf*)(Qbf + (size_t)(b * NPIX + nt * 16 + lane) * DK);

    v8f s = {};
    s = __builtin_amdgcn_wmma_f32_16x16x32_bf16(false, qa.v, false, kb.v,
                                                (short)0, s, false, false);
    // Per-column reduction: 8 rows in this lane + 8 rows in lane^16.
    float tmax = s[0];
    #pragma unroll
    for (int r = 1; r < 8; ++r) tmax = fmaxf(tmax, s[r]);
    tmax = fmaxf(tmax, __shfl_xor(tmax, 16, 32));
    const float nm = fmaxf(run_max, tmax);
    run_sum *= __expf(run_max - nm);
    float ps = 0.f;
    #pragma unroll
    for (int r = 0; r < 8; ++r) ps += __expf(s[r] - nm);
    ps += __shfl_xor(ps, 16, 32);
    run_sum += ps;
    run_max = nm;
  }
  if (lane < 16) {
    maxbuf[b * NPIX + m_base + lane] = run_max;
    sumbuf[b * NPIX + m_base + lane] = run_sum;
  }
}

// ---------------------------------------------------------------------------
// Kernel 3: output. One block per 32-query tile (two 16-row WMMA tiles) x all
// 64 channels. Per 64-key iteration: wave w computes S/P for its own 16-key
// group against BOTH query tiles (2 QK WMMAs, fully parallel across waves),
// then every wave consumes all 64 keys with 4 accumulated AV WMMAs.
// ---------------------------------------------------------------------------
#define PL_STRIDE 72   // 64 keys + pad (16 distinct bank phases across rows)

__global__ __launch_bounds__(128) void attnout_kernel(
    const float* __restrict__ x,
    const unsigned short* __restrict__ Qbf,
    const unsigned short* __restrict__ Kbf,
    const unsigned short* __restrict__ Vbf,
    const float* __restrict__ maxbuf, const float* __restrict__ sumbuf,
    float* __restrict__ out)
{
  __shared__ unsigned short P_lds[32 * PL_STRIDE];  // [query n (32)][key m_local (64)]

  const int lane = threadIdx.x & 31;
  const int wave = threadIdx.x >> 5;
  const int b  = blockIdx.x / (NPIX / 32);
  const int n0 = (blockIdx.x % (NPIX / 32)) * 32;
  const int l15 = lane & 15;
  const bool lo = lane < 16;
  const int rbase = lo ? 0 : 8;        // C/D row half for this lane
  const int mo    = lo ? 0 : 8;        // A-layout K offset for this lane half

  // Q A-fragments for the two query sub-tiles.
  BF16Frag qa[2];
  #pragma unroll
  for (int t = 0; t < 2; ++t) {
    #pragma unroll
    for (int i = 0; i < 16; ++i) qa[t].us[i] = 0;
    if (lo)
      qa[t].v = *(const v16bf*)(Qbf + (size_t)(b * NPIX + n0 + t * 16 + lane) * DK);
  }

  v8f acc0 = {}, acc1 = {};
  const int c_row = wave * 16 + l15;   // A-matrix row = output channel
  const unsigned short* vrow = Vbf + (size_t)(b * CCH + c_row) * NPIX;
  const float* mxb = maxbuf + b * NPIX;
  const float* smb = sumbuf + b * NPIX;

  for (int mt = 0; mt < NPIX / 64; ++mt) {
    const int m0 = mt * 64;
    const int mk = m0 + wave * 16;     // this wave's 16-key S/P group

    // --- S/P for key group [mk, mk+16), both query tiles, all waves parallel.
    BF16Frag kb;
    #pragma unroll
    for (int i = 0; i < 16; ++i) kb.us[i] = 0;
    if (lo)
      kb.v = *(const v16bf*)(Kbf + (size_t)(b * NPIX + mk + lane) * DK);

    const float mx  = mxb[mk + l15];              // this lane's key column stats
    const float inv = 1.f / smb[mk + l15];

    #pragma unroll
    for (int t = 0; t < 2; ++t) {
      v8f s = {};
      s = __builtin_amdgcn_wmma_f32_16x16x32_bf16(false, qa[t].v, false, kb.v,
                                                  (short)0, s, false, false);
      #pragma unroll
      for (int r = 0; r < 8; ++r) {
        const float p = __expf(s[r] - mx) * inv;
        P_lds[(t * 16 + rbase + r) * PL_STRIDE + wave * 16 + l15] = f32_to_bf16(p);
      }
    }
    __syncthreads();

    // --- AV: out[c, n] += sum_m V[c,m] * P[n,m] over the 64 keys.
    #pragma unroll
    for (int ch = 0; ch < 2; ++ch) {   // two 32-key K-chunks
      const int mc = m0 + ch * 32;
      BF16Frag va;                     // V rows c, K = mc + {mo..mo+7, 16+mo..}
      va.h[0] = *(const v8bf*)(vrow + mc + mo);
      va.h[1] = *(const v8bf*)(vrow + mc + 16 + mo);

      #pragma unroll
      for (int t = 0; t < 2; ++t) {    // two query sub-tiles
        BF16Frag pb;                   // P^T: column n = l15, K = {mo..mo+7, 16+mo..}
        const unsigned short* prow =
            &P_lds[(t * 16 + l15) * PL_STRIDE + ch * 32 + (lo ? 0 : 8)];
        pb.h[0] = *(const v8bf*)(prow);
        pb.h[1] = *(const v8bf*)(prow + 16);
        if (t == 0)
          acc0 = __builtin_amdgcn_wmma_f32_16x16x32_bf16(false, va.v, false, pb.v,
                                                         (short)0, acc0, false, false);
        else
          acc1 = __builtin_amdgcn_wmma_f32_16x16x32_bf16(false, va.v, false, pb.v,
                                                         (short)0, acc1, false, false);
      }
    }

    if (mt + 1 < NPIX / 64)
      __builtin_prefetch(vrow + m0 + 64 + mo, 0, 0);        // global_prefetch_b8

    __syncthreads();                                        // P_lds reuse guard
  }

  // C/D layout: lane holds column n, rows M = rbase + r.
  #pragma unroll
  for (int r = 0; r < 8; ++r) {
    const int c = wave * 16 + rbase + r;
    const size_t i0 = (size_t)(b * CCH + c) * NPIX + (n0 + l15);
    out[i0]      = acc0[r] + x[i0];                         // residual add
    out[i0 + 16] = acc1[r] + x[i0 + 16];
  }
}

// ---------------------------------------------------------------------------
extern "C" void kernel_launch(void* const* d_in, const int* in_sizes, int n_in,
                              void* d_out, int out_size, void* d_ws, size_t ws_size,
                              hipStream_t stream) {
  const float* x  = (const float*)d_in[0];
  const float* Wq = (const float*)d_in[1];
  const float* bq = (const float*)d_in[2];
  const float* Wk = (const float*)d_in[3];
  const float* bk = (const float*)d_in[4];
  const float* Wv = (const float*)d_in[5];
  const float* bv = (const float*)d_in[6];
  float* out = (float*)d_out;

  // Workspace layout (all L2-resident): Q 1MB | K 1MB | V 2MB | max 64KB | sum 64KB
  char* ws = (char*)d_ws;
  unsigned short* Qbf = (unsigned short*)(ws);
  unsigned short* Kbf = (unsigned short*)(ws + (1u << 20));
  unsigned short* Vbf = (unsigned short*)(ws + (2u << 20));
  float* maxbuf = (float*)(ws + (4u << 20));
  float* sumbuf = (float*)(ws + (4u << 20) + (size_t)BATCH * NPIX * 4);

  hipLaunchKernelGGL(proj_kernel, dim3(BATCH * NPIX / 256), dim3(256), 0, stream,
                     x, Wq, bq, Wk, bk, Wv, bv, Qbf, Kbf, Vbf);
  hipLaunchKernelGGL(stats_kernel, dim3(BATCH * (NPIX / 64)), dim3(128), 0, stream,
                     Qbf, Kbf, maxbuf, sumbuf);
  hipLaunchKernelGGL(attnout_kernel, dim3(BATCH * (NPIX / 32)), dim3(128), 0, stream,
                     x, Qbf, Kbf, Vbf, maxbuf, sumbuf, out);
}
